// WindowFrequencyModulation_FMFFN_1812476199057
// MI455X (gfx1250) — matrix operand order repaired and checked by reading, not compiled, not run on hardware
//
#include <hip/hip_runtime.h>
#include <hip/hip_bf16.h>

// WindowFrequencyModulation on MI455X (gfx1250):
//   rfft2(8x8,ortho) -> * W_c -> irfft2(8x8,ortho)  ==  per-channel 64x64 circulant GEMM.
// Memory-bound: 403 MB HBM traffic -> ~17 us floor @ 23.3 TB/s; 6.4 GFLOP is trivial
// for v_wmma_f32_16x16x32_f16 (f16 A/B, f32 accumulate). A_c matrices (3 MB f16 total)
// are precomputed into WMMA A-fragment order in d_ws and kept L2-resident; the pure
// streaming x/out traffic is tagged non-temporal (ISA 7.3 TH=NT) so it does not evict them.

typedef __attribute__((ext_vector_type(16))) _Float16 v16h;
typedef __attribute__((ext_vector_type(8)))  float    v8f;
typedef __attribute__((ext_vector_type(4)))  float    v4f;   // native vec4 for NT load/store builtins

#define NP    8      // window size P
#define NC    384    // channels
#define NB    8      // batch
#define IMG   128    // H = W
#define NW    16     // windows per image dimension (128/8)

// ---------------------------------------------------------------------------
// Kernel 1: per channel c, build h_c[8][8] (spatial circular-conv kernel) from the
// complex half-spectrum weight [8][5][C][2], then emit the 64x64 circulant operator
// directly in v_wmma_f32_16x16x32_f16 A-fragment order:
//   afrag[c][kstep(2)][mtile(4)][lane(32)][t(16)]  (f16)  => 8 KB per channel.
// A-fragment k order used here (contiguous, same as the X B-fragment): lane holds row M = lane%16;
//   laneLo: t=0..15 -> K=t (0..15)
//   laneHi: t=0..15 -> K=t+16 (16..31)
// plus kstep*32.
// ---------------------------------------------------------------------------
__global__ __launch_bounds__(256) void wfm_build_afrag(const float* __restrict__ cw,
                                                       _Float16* __restrict__ afrag)
{
    __shared__ float h[64];
    const int c   = blockIdx.x;
    const int tid = threadIdx.x;

    if (tid < 64) {
        const int d1 = tid >> 3, d2 = tid & 7;
        float s = 0.0f;
        #pragma unroll
        for (int k1 = 0; k1 < 8; ++k1) {
            #pragma unroll
            for (int k2 = 0; k2 < 8; ++k2) {
                // Hermitian extension of the rfft2 half-spectrum along k2.
                float re, im;
                if (k2 <= 4) {
                    const float* wp = cw + ((size_t)(k1 * 5 + k2) * NC + c) * 2;
                    re = wp[0]; im = wp[1];
                } else {
                    const int k1f = (8 - k1) & 7;
                    const int k2f = 8 - k2;
                    const float* wp = cw + ((size_t)(k1f * 5 + k2f) * NC + c) * 2;
                    re = wp[0]; im = -wp[1];
                }
                const float ang = 0.78539816339744831f * (float)((k1 * d1 + k2 * d2) & 7);
                float sn, cn;
                __sincosf(ang, &sn, &cn);
                s += re * cn - im * sn;   // Re(Wfull * e^{+i ang})
            }
        }
        h[tid] = s * (1.0f / 64.0f);      // ortho norm both ways: 1/sqrt(64) twice
    }
    __syncthreads();

    // 256 threads == 2 ksteps * 4 mtiles * 32 lanes: each thread writes one 16-half fragment row.
    const int kstep  = tid >> 7;
    const int mtile  = (tid >> 5) & 3;
    const int lane   = tid & 31;
    const int laneHi = lane >> 4;
    const int M      = lane & 15;            // output row within 16-row M-tile
    const int p1     = mtile * 2 + (M >> 3); // window row of output element
    const int p2     = M & 7;                // window col of output element

    v16h frag;
    #pragma unroll
    for (int t = 0; t < 16; ++t) {
        const int K  = kstep * 32 + 16 * laneHi + t; // input index q = q1*8+q2
        const int q1 = (K >> 3) & 7;
        const int q2 = K & 7;
        frag[t] = (_Float16)h[(((p1 - q1) & 7) << 3) | ((p2 - q2) & 7)];
    }
    _Float16* dst = afrag + ((((size_t)c * 8 + (kstep * 4 + mtile)) * 32 + lane) << 4);
    *(v16h*)dst = frag;   // 32B aligned (d_ws is 256B aligned; offsets are 32B multiples)
}

// ---------------------------------------------------------------------------
// Kernel 2: apply Y[64 x 16win] = A_c[64x64] * X[64 x 16win] per (b, c, w1-stripe).
// Block = (c, b), 256 threads = 8 waves; wave w handles stripes w1 = 2w, 2w+1.
// B-fragment (16-bit, 32x16): lane col N = lane%16 = window w2;
//   laneLo holds K=0..15 (+32*kstep), laneHi K=16..31 -> two 8-float image rows per lane,
//   loaded as vec4 pairs: 16 lanes cover one full 512B image row (fully coalesced).
// D (16x16 f32, 8 VGPRs): lane's 8 values are p2=0..7 of row p1 = 2*mtile + laneHi
//   -> two global_store_b128 per M-tile, fully coalesced.
// x loads / out stores are non-temporal (read-once / write-once streaming);
// A-fragment loads stay RT so the 3 MB operator tensor persists in the 192 MB L2.
// ---------------------------------------------------------------------------
__global__ __launch_bounds__(256) void wfm_apply(const float* __restrict__ x,
                                                 const _Float16* __restrict__ afrag,
                                                 float* __restrict__ out)
{
    const int c      = blockIdx.x;
    const int b      = blockIdx.y;
    const int wave   = threadIdx.x >> 5;
    const int lane   = threadIdx.x & 31;
    const int laneHi = lane >> 4;
    const int w2     = lane & 15;

    const size_t plane = ((size_t)b * NC + c) * (IMG * IMG);
    const float* xc = x + plane;
    float*       oc = out + plane;

    // Load the 8 A fragments for this channel (coalesced 32B/lane; hits L2 after first use).
    v16h a[4][2];
    const _Float16* ab = afrag + (((size_t)c * 8) << 9);   // c * 8*32*16
    #pragma unroll
    for (int ks = 0; ks < 2; ++ks) {
        #pragma unroll
        for (int m = 0; m < 4; ++m) {
            a[m][ks] = *(const v16h*)(ab + ((size_t)((ks * 4 + m) * 32 + lane) << 4));
        }
    }

    #pragma unroll
    for (int s = 0; s < 2; ++s) {
        const int w1 = wave * 2 + s;

        // Gather X fragments straight from global (f32 -> f16), non-temporal.
        v16h bf[2];
        #pragma unroll
        for (int ks = 0; ks < 2; ++ks) {
            const int q1b = laneHi * 2 + ks * 4;
            const v4f* r0 = (const v4f*)(xc + (size_t)(w1 * 8 + q1b) * IMG + w2 * 8);
            const v4f* r1 = (const v4f*)(xc + (size_t)(w1 * 8 + q1b + 1) * IMG + w2 * 8);
            const v4f x0 = __builtin_nontemporal_load(r0);
            const v4f x1 = __builtin_nontemporal_load(r0 + 1);
            const v4f y0 = __builtin_nontemporal_load(r1);
            const v4f y1 = __builtin_nontemporal_load(r1 + 1);
            v16h t;
            t[0]  = (_Float16)x0.x; t[1]  = (_Float16)x0.y; t[2]  = (_Float16)x0.z; t[3]  = (_Float16)x0.w;
            t[4]  = (_Float16)x1.x; t[5]  = (_Float16)x1.y; t[6]  = (_Float16)x1.z; t[7]  = (_Float16)x1.w;
            t[8]  = (_Float16)y0.x; t[9]  = (_Float16)y0.y; t[10] = (_Float16)y0.z; t[11] = (_Float16)y0.w;
            t[12] = (_Float16)y1.x; t[13] = (_Float16)y1.y; t[14] = (_Float16)y1.z; t[15] = (_Float16)y1.w;
            bf[ks] = t;
        }

        // 4 M-tiles x (K=64 as 2 chained k-steps): 8 WMMAs per stripe.
        #pragma unroll
        for (int m = 0; m < 4; ++m) {
            v8f acc = {};
            acc = __builtin_amdgcn_wmma_f32_16x16x32_f16(false, a[m][0], false, bf[0],
                                                         (short)0, acc, false, false);
            acc = __builtin_amdgcn_wmma_f32_16x16x32_f16(false, a[m][1], false, bf[1],
                                                         (short)0, acc, false, false);
            const int p1 = m * 2 + laneHi;
            float* orow = oc + (size_t)(w1 * 8 + p1) * IMG + w2 * 8;
            const v4f d0 = { acc[0], acc[1], acc[2], acc[3] };
            const v4f d1 = { acc[4], acc[5], acc[6], acc[7] };
            __builtin_nontemporal_store(d0, (v4f*)orow);
            __builtin_nontemporal_store(d1, (v4f*)orow + 1);
        }
    }
}

// ---------------------------------------------------------------------------
extern "C" void kernel_launch(void* const* d_in, const int* in_sizes, int n_in,
                              void* d_out, int out_size, void* d_ws, size_t ws_size,
                              hipStream_t stream) {
    const float* x  = (const float*)d_in[0];            // [8,384,128,128] f32
    const float* cw = (const float*)d_in[1];            // [8,5,384,2] f32
    float*       out = (float*)d_out;                   // [8,384,128,128] f32
    _Float16*    afrag = (_Float16*)d_ws;               // needs 384*8*32*16*2 = 3,145,728 B

    (void)in_sizes; (void)n_in; (void)out_size; (void)ws_size;

    // 1) Build per-channel operators in WMMA fragment order (tiny: 384 blocks).
    wfm_build_afrag<<<NC, 256, 0, stream>>>(cw, afrag);

    // 2) Streaming batched GEMM: (channels x batches) blocks, 8 waves each.
    wfm_apply<<<dim3(NC, NB), 256, 0, stream>>>(x, afrag, out);
}
